// DMPNN_54949811585617
// MI455X (gfx1250) — matrix operand order repaired and checked by reading, not compiled, not run on hardware
//
#include <hip/hip_runtime.h>

// ---------------------------------------------------------------------------
// D-MPNN forward on MI455X (gfx1250), wave32 + v_wmma_f32_16x16x32_bf16.
// Memory-bound problem: edge tensor h [E=800000, H=128] dominates traffic.
// h kept in bf16; all segment sums accumulated in f32 via global atomics
// fused into the WMMA epilogues.
// ---------------------------------------------------------------------------

typedef __attribute__((ext_vector_type(16))) __bf16 v16bf;
typedef __attribute__((ext_vector_type(8)))  float  v8f;

#define N_NODES 50000
#define N_UND   400000
#define E_DIR   (2 * N_UND)
#define AD      64
#define BD      16
#define H       128

// Workspace layout (bytes). Total ~= 256.1 MB.
static constexpr size_t OFF_H   = 0;                                   // E*H bf16
static constexpr size_t OFF_AGG = OFF_H   + (size_t)E_DIR * H * 2;     // N*H f32
static constexpr size_t OFF_NM  = OFF_AGG + (size_t)N_NODES * H * 4;   // N*H f32
static constexpr size_t OFF_PWI = OFF_NM  + (size_t)N_NODES * H * 4;   // 3*8*512 bf16
static constexpr size_t OFF_PWM = OFF_PWI + (size_t)3 * 4096 * 2;
static constexpr size_t OFF_PWA = OFF_PWM + (size_t)4 * 4096 * 2;

// --------------------------- fragment loaders ------------------------------

// A-matrix 16x32 bf16 fragment from an LDS tile with row stride `kpad` halves.
// ISA layout: lanes 0-15 hold row M=lane, K = k0+{0..7} then k0+{16..23};
// lanes 16-31 hold row M=lane-16, K = k0+{8..15} then k0+{24..31}.
__device__ __forceinline__ v16bf load_a_lds(const __bf16* A, int kpad, int k0, int lane) {
  const __bf16* p = A + (lane & 15) * kpad + k0 + ((lane >> 4) << 3);
  union { uint4 u[2]; v16bf v; } cvt;
  cvt.u[0] = *(const uint4*)p;        // 8 halves (16B aligned)
  cvt.u[1] = *(const uint4*)(p + 16); // 8 halves
  return cvt.v;
}

// bf16 -> f32 from packed pair (low/high 16 bits); bf16->f32 is bits<<16.
__device__ __forceinline__ float bf16lo(unsigned u) { return __uint_as_float(u << 16); }
__device__ __forceinline__ float bf16hi(unsigned u) { return __uint_as_float(u & 0xffff0000u); }

// ------------------------------ prep kernels -------------------------------

// Pack a [Krows x 128] f32 weight into bf16 B-fragments:
// element i = ((kt*8 + ntile)*32 + lane)*16 + h  holds  W[k, col] with
// k = kt*32 + (h>>3)*16 + (lane>>4)*8 + (h&7) (B layout: lanes 0-15 -> K 0..7,16..23;
// lanes 16-31 -> K 8..15,24..31 of the tile), col = ntile*16 + (lane&15). Zero-pad k >= Krows.
__device__ __forceinline__ void pack_one(const float* __restrict__ W, int Krows, int ktiles,
                                         __bf16* __restrict__ out, int t, int nthreads) {
  int total = ktiles * 4096; // 8 ntiles * 32 lanes * 16 halves
  for (int i = t; i < total; i += nthreads) {
    int h     = i & 15;
    int lane  = (i >> 4) & 31;
    int ntile = (i >> 9) & 7;
    int kt    = i >> 12;
    int k     = kt * 32 + ((h & 8) << 1) + ((lane >> 4) << 3) + (h & 7);
    int col   = (ntile << 4) + (lane & 15);
    out[i] = (k < Krows) ? (__bf16)W[(size_t)k * H + col] : (__bf16)0.0f;
  }
}

__global__ __launch_bounds__(256) void pack_weights_kernel(
    const float* __restrict__ Wi, const float* __restrict__ Wm, const float* __restrict__ Wa,
    __bf16* __restrict__ pWi, __bf16* __restrict__ pWm, __bf16* __restrict__ pWa) {
  int t = blockIdx.x * blockDim.x + threadIdx.x;
  int n = gridDim.x * blockDim.x;
  pack_one(Wi,  80, 3, pWi, t, n);
  pack_one(Wm, 128, 4, pWm, t, n);
  pack_one(Wa, 192, 6, pWa, t, n);
}

__global__ __launch_bounds__(256) void zero_kernel(float* __restrict__ agg,
                                                   float* __restrict__ nm,
                                                   float* __restrict__ out) {
  size_t i = (size_t)blockIdx.x * blockDim.x + threadIdx.x;
  size_t stride = (size_t)gridDim.x * blockDim.x;
  size_t total = (size_t)N_NODES * H;
  for (size_t k = i; k < total; k += stride) { agg[k] = 0.0f; nm[k] = 0.0f; }
  if (i < H) out[i] = 0.0f;
}

// ------------------------- stage 1: edge init GEMM -------------------------
// h[e,:] = relu([x[src[e]] | efeat[e]] @ W_i); fused scatter agg[dst[e],:] += h.
// Block = 16 edges, 256 threads = 8 waves, wave w owns columns [16w,16w+16).
__global__ __launch_bounds__(256) void edge_init_kernel(
    const float* __restrict__ xn, const float* __restrict__ xe,
    const int* __restrict__ src, const int* __restrict__ dst,
    const __bf16* __restrict__ pWi, __bf16* __restrict__ hout,
    float* __restrict__ agg) {
  __shared__ __align__(16) __bf16 As[16 * 96]; // K padded 80 -> 96
  __shared__ int sdst[16];
  const int e0  = blockIdx.x * 16;
  const int tid = threadIdx.x;
  if (tid < 16) sdst[tid] = dst[e0 + tid];
  for (int i = tid; i < 16 * 96; i += 256) {
    int row = i / 96, col = i - row * 96;
    float v;
    if (col < AD)            v = xn[(size_t)src[e0 + row] * AD + col];
    else if (col < AD + BD)  v = xe[(size_t)(e0 + row) * BD + (col - AD)];
    else                     v = 0.0f;
    As[i] = (__bf16)v;
  }
  __syncthreads();

  const int wv = tid >> 5, lane = tid & 31;
  v8f acc = {};
#pragma unroll
  for (int kt = 0; kt < 3; ++kt) {
    v16bf a = load_a_lds(As, 96, kt * 32, lane);
    v16bf b = ((const v16bf*)pWi)[(kt * 8 + wv) * 32 + lane];
    acc = __builtin_amdgcn_wmma_f32_16x16x32_bf16(false, a, false, b, (short)0, acc, false, false);
  }
  const int col = wv * 16 + (lane & 15);
  const int mb  = (lane >> 4) * 8; // VGPR r: lanes 0-15 -> M=r, lanes 16-31 -> M=r+8
#pragma unroll
  for (int r = 0; r < 8; ++r) {
    int row = mb + r;
    float v = fmaxf(acc[r], 0.0f);
    hout[(size_t)(e0 + row) * H + col] = (__bf16)v;
    atomicAdd(&agg[(size_t)sdst[row] * H + col], v);
  }
}

// ----------------------- stage 2: edge message GEMM ------------------------
// m[e,:] = agg[src[e],:] - h[rev[e],:];  h2 = relu(m @ W_m);
// fused scatter node_m[dst[e],:] += h2.
__global__ __launch_bounds__(256) void edge_msg_kernel(
    const float* __restrict__ agg, const __bf16* __restrict__ hin,
    const int* __restrict__ src, const int* __restrict__ dst,
    const int* __restrict__ rev, const __bf16* __restrict__ pWm,
    float* __restrict__ node_m) {
  __shared__ __align__(16) __bf16 As[16 * 128];
  __shared__ int sdst[16];
  const int e0  = blockIdx.x * 16;
  const int tid = threadIdx.x;
  if (tid < 16) sdst[tid] = dst[e0 + tid];
  {
    // Each thread builds one row x 8-col chunk of A (16 rows * 16 chunks).
    int row = tid >> 4;
    int c8  = (tid & 15) * 8;
    int e   = e0 + row;
    int s = src[e], rv = rev[e];
    const float4* ap = (const float4*)(agg + (size_t)s * H + c8);
    float4 a0 = ap[0], a1 = ap[1];
    uint4 hb = *(const uint4*)(hin + (size_t)rv * H + c8);
    __bf16* o = As + row * H + c8;
    o[0] = (__bf16)(a0.x - bf16lo(hb.x)); o[1] = (__bf16)(a0.y - bf16hi(hb.x));
    o[2] = (__bf16)(a0.z - bf16lo(hb.y)); o[3] = (__bf16)(a0.w - bf16hi(hb.y));
    o[4] = (__bf16)(a1.x - bf16lo(hb.z)); o[5] = (__bf16)(a1.y - bf16hi(hb.z));
    o[6] = (__bf16)(a1.z - bf16lo(hb.w)); o[7] = (__bf16)(a1.w - bf16hi(hb.w));
  }
  __syncthreads();

  const int wv = tid >> 5, lane = tid & 31;
  v8f acc = {};
#pragma unroll
  for (int kt = 0; kt < 4; ++kt) {
    v16bf a = load_a_lds(As, 128, kt * 32, lane);
    v16bf b = ((const v16bf*)pWm)[(kt * 8 + wv) * 32 + lane];
    acc = __builtin_amdgcn_wmma_f32_16x16x32_bf16(false, a, false, b, (short)0, acc, false, false);
  }
  const int col = wv * 16 + (lane & 15);
  const int mb  = (lane >> 4) * 8;
#pragma unroll
  for (int r = 0; r < 8; ++r) {
    float v = fmaxf(acc[r], 0.0f);
    atomicAdd(&node_m[(size_t)sdst[mb + r] * H + col], v);
  }
}

// --------------------------- stage 3: readout ------------------------------
// node_emb = relu([x | node_m] @ W_a); out = sum over nodes.
__global__ __launch_bounds__(256) void readout_kernel(
    const float* __restrict__ xn, const float* __restrict__ node_m,
    const __bf16* __restrict__ pWa, float* __restrict__ out) {
  __shared__ __align__(16) __bf16 As[16 * 192];
  __shared__ float osum[H];
  const int n0  = blockIdx.x * 16;
  const int tid = threadIdx.x;
  for (int i = tid; i < 16 * 192; i += 256) {
    int row = i / 192, col = i - row * 192;
    float v = (col < AD) ? xn[(size_t)(n0 + row) * AD + col]
                         : node_m[(size_t)(n0 + row) * H + (col - AD)];
    As[i] = (__bf16)v;
  }
  if (tid < H) osum[tid] = 0.0f;
  __syncthreads();

  const int wv = tid >> 5, lane = tid & 31;
  v8f acc = {};
#pragma unroll
  for (int kt = 0; kt < 6; ++kt) {
    v16bf a = load_a_lds(As, 192, kt * 32, lane);
    v16bf b = ((const v16bf*)pWa)[(kt * 8 + wv) * 32 + lane];
    acc = __builtin_amdgcn_wmma_f32_16x16x32_bf16(false, a, false, b, (short)0, acc, false, false);
  }
  // Reduce 8 rows held by this lane (relu first), then LDS-combine per column.
  float s = 0.0f;
#pragma unroll
  for (int r = 0; r < 8; ++r) s += fmaxf(acc[r], 0.0f);
  atomicAdd(&osum[wv * 16 + (lane & 15)], s);
  __syncthreads();
  if (tid < H) atomicAdd(&out[tid], osum[tid]);
}

// ------------------------------ launcher -----------------------------------

extern "C" void kernel_launch(void* const* d_in, const int* in_sizes, int n_in,
                              void* d_out, int out_size, void* d_ws, size_t ws_size,
                              hipStream_t stream) {
  (void)in_sizes; (void)n_in; (void)out_size; (void)ws_size;
  const float* xn  = (const float*)d_in[0];   // [N, 64]
  const float* xe  = (const float*)d_in[1];   // [E, 16]
  const float* Wi  = (const float*)d_in[2];   // [80, 128]
  const float* Wm  = (const float*)d_in[3];   // [128, 128]
  const float* Wa  = (const float*)d_in[4];   // [192, 128]
  const int*   ei  = (const int*)d_in[5];     // [2, E]
  const int*   src = ei;
  const int*   dst = ei + E_DIR;
  const int*   rev = (const int*)d_in[6];     // [E]

  char*   ws   = (char*)d_ws;
  __bf16* hbuf = (__bf16*)(ws + OFF_H);
  float*  agg  = (float*)(ws + OFF_AGG);
  float*  nm   = (float*)(ws + OFF_NM);
  __bf16* pWi  = (__bf16*)(ws + OFF_PWI);
  __bf16* pWm  = (__bf16*)(ws + OFF_PWM);
  __bf16* pWa  = (__bf16*)(ws + OFF_PWA);
  float*  out  = (float*)d_out;

  hipLaunchKernelGGL(pack_weights_kernel, dim3(52), dim3(256), 0, stream,
                     Wi, Wm, Wa, pWi, pWm, pWa);
  hipLaunchKernelGGL(zero_kernel, dim3(2048), dim3(256), 0, stream, agg, nm, out);
  hipLaunchKernelGGL(edge_init_kernel, dim3(E_DIR / 16), dim3(256), 0, stream,
                     xn, xe, src, dst, pWi, hbuf, agg);
  hipLaunchKernelGGL(edge_msg_kernel, dim3(E_DIR / 16), dim3(256), 0, stream,
                     agg, hbuf, src, dst, rev, pWm, nm);
  hipLaunchKernelGGL(readout_kernel, dim3(N_NODES / 16), dim3(256), 0, stream,
                     xn, nm, pWa, out);
}
